// MultiGraphConvLayer_90305982366173
// MI455X (gfx1250) — matrix operand order repaired and checked by reading, not compiled, not run on hardware
//
#include <hip/hip_runtime.h>
#include <hip/hip_bf16.h>

// MultiGraphConvLayer for MI455X (gfx1250).
// ~77 GFLOP, ~92 MB working set (fits 192MB L2) -> compute bound.
// All GEMM stages run v_wmma_f32_16x16x32_f16 with f32 accumulation; operand
// tiles are DMA'd global->LDS with the Tensor Data Mover (double-buffered,
// TENSORcnt-synchronized), with TDM LDS-padding (80B row pitch) so the
// ds_load_b128 fragment reads are bank-conflict free.
// Block tile 128x128, 8 waves, 32x64 C per wave (8 WMMAs per barrier pair).
// TDM descriptors are built in uniform control flow (pure SALU) and the
// k-loop is unrolled 2x so buffer offsets are immediates.
// adj@X + X is folded to (adj + I)@X at convert time.
// Workspace requirement: ~115 MB.

namespace {

constexpr int kH = 3, kL = 2, kB = 16, kS = 512, kD = 512;
constexpr int kF = kH * kL * kD;          // 3072
constexpr int kM = kB * kS;               // 8192 flattened rows

typedef __attribute__((ext_vector_type(16))) _Float16 v16h;
typedef __attribute__((ext_vector_type(8)))  _Float16 v8h;
typedef __attribute__((ext_vector_type(8)))  float    v8f;
typedef __attribute__((ext_vector_type(4)))  unsigned int v4u;
typedef __attribute__((ext_vector_type(8)))  int v8i;
typedef __attribute__((ext_vector_type(4)))  int v4i;

union FragH { v16h v; v8h h[2]; };

// LDS geometry: tile rows are 64B of payload + 16B TDM pad -> 80B pitch.
constexpr int kRowPitch = 80;
constexpr unsigned kTileBytes = 128u * kRowPitch;   // 10240B per 128-row tile
constexpr unsigned kLdsA   = 0;                     // A tiles x2
constexpr unsigned kLdsB   = 2 * kTileBytes;        // B tiles x2
constexpr unsigned kLdsTot = 4 * kTileBytes;        // 40960B

typedef __attribute__((address_space(3))) char as3_char;
__device__ __forceinline__ unsigned lds_off(void* p) {
  return (unsigned)(unsigned long long)(as3_char*)p;
}

// ---- Tensor Data Mover descriptors (2D tile, data_size=8B, 64B rows,
// LDS pad 16B per 64B -> 80B pitch). Built as wave-uniform scalars.
__device__ __forceinline__ v4u tdm_g0(unsigned lds_byte_off, const void* gptr) {
  const unsigned long long ga = (unsigned long long)(uintptr_t)gptr;
  v4u g0;
  g0[0] = 1u;                                               // count=1
  g0[1] = lds_byte_off;                                     // lds_addr
  g0[2] = (unsigned)(ga & 0xffffffffu);                     // global_addr[31:0]
  g0[3] = (unsigned)((ga >> 32) & 0x1ffffffu) | (2u << 30); // addr[56:32], type=2
  return g0;
}

__device__ __forceinline__ v8i tdm_g1(int rows, unsigned stride_u) {
  v8i g1;
  // data_size=3 (8B) | pad_enable | pad_interval=16dw(code3) | pad_amount=4dw(code3)
  g1[0] = (int)((3u << 16) | (1u << 20) | (3u << 22) | (3u << 25));
  g1[1] = (int)0xffff0000u;                  // barrier_addr=0, tensor_dim0 lo
  g1[2] = (int)0xffff7fffu;                  // tensor_dim0 hi, tensor_dim1 lo
  g1[3] = (int)0x00087fffu;                  // tensor_dim1 hi, tile_dim0=8
  g1[4] = rows & 0xffff;                     // tile_dim1=rows, tile_dim2=0
  g1[5] = (int)stride_u;                     // tensor_dim0_stride[31:0]
  g1[6] = 0;
  g1[7] = 0;
  return g1;
}

__device__ __forceinline__ void tdm_issue(v4u g0, v8i g1) {
  const v4i gz4 = {0, 0, 0, 0};
  const v8i gz8 = {0, 0, 0, 0, 0, 0, 0, 0};
  __builtin_amdgcn_tensor_load_to_lds(g0, g1, gz4, gz4, gz8, 0);
}

// A fragment 16x32 from padded LDS tile (rows at 80B pitch).
// A layout (contiguous k, matches B): lanes 0-15 rows M=0..15 K=0..15; lanes 16-31 K=16..31.
__device__ __forceinline__ v16h lds_frag_A(const char* base, int row0) {
  const int lane = threadIdx.x & 31;
  const int r  = row0 + (lane & 15);
  const int kb = (lane >> 4) << 4;                 // 0 or 16 halves
  const char* p = base + r * kRowPitch + kb * 2;
  FragH f;
  f.h[0] = *(const v8h*)(p);
  f.h[1] = *(const v8h*)(p + 16);
  return f.v;
}

// B fragment 32x16: source rows are B's columns; lanes 0-15 K=0..15, 16-31 K=16..31.
__device__ __forceinline__ v16h lds_frag_B(const char* base, int row0) {
  const int lane = threadIdx.x & 31;
  const int r  = row0 + (lane & 15);
  const int kb = (lane >> 4) << 4;                 // 0 or 16 halves
  const char* p = base + r * kRowPitch + kb * 2;
  FragH f;
  f.h[0] = *(const v8h*)(p);
  f.h[1] = *(const v8h*)(p + 16);
  return f.v;
}

__device__ __forceinline__ v8f wmma_f16(v16h a, v16h b, v8f c) {
  return __builtin_amdgcn_wmma_f32_16x16x32_f16(false, a, false, b,
                                                (short)0, c, false, false);
}

// ---- TDM-fed GEMM core: block tile 128(M)x128(N), 8 waves, 32x64 C per wave.
// Requires nk even. Ag/Bg point at (m_block|n_block, k=0); lda == ldb in all
// users so A and B share the loop-invariant g1 descriptor.
__device__ __forceinline__ void gemm_tdm_core(const _Float16* Ag,
                                              const _Float16* Bg, int ld,
                                              int nk, char* sm,
                                              v8f acc[2][4]) {
  const int tid  = threadIdx.x;
  const int wave = tid >> 5;
  const int m0w  = (wave >> 1) << 5;    // 0..96
  const int n0w  = (wave & 1) << 6;     // 0 or 64
  const unsigned base = lds_off(sm);
  const v8i g1 = tdm_g1(128, (unsigned)(ld >> 2));

  auto compute = [&](unsigned aoff, unsigned boff) {
    const char* As = sm + aoff;
    const char* Bs = sm + boff;
    v16h a0 = lds_frag_A(As, m0w);
    v16h a1 = lds_frag_A(As, m0w + 16);
    v16h bfr[4];
#pragma unroll
    for (int ni = 0; ni < 4; ++ni) bfr[ni] = lds_frag_B(Bs, n0w + ni * 16);
#pragma unroll
    for (int ni = 0; ni < 4; ++ni) {
      acc[0][ni] = wmma_f16(a0, bfr[ni], acc[0][ni]);
      acc[1][ni] = wmma_f16(a1, bfr[ni], acc[1][ni]);
    }
  };

  // Stage 0 -> buffer 0 (descriptors formed uniformly; branch holds only DMA).
  {
    const v4u a0 = tdm_g0(base + kLdsA, Ag);
    const v4u b0 = tdm_g0(base + kLdsB, Bg);
    if (tid < 32) { tdm_issue(a0, g1); tdm_issue(b0, g1); }
  }
  for (int kb = 0; kb < nk; kb += 2) {
    // ---- even step: prefetch stage kb+1 -> buffer 1, consume buffer 0
    {
      const v4u a0 = tdm_g0(base + kLdsA + kTileBytes, Ag + (size_t)(kb + 1) * 32);
      const v4u b0 = tdm_g0(base + kLdsB + kTileBytes, Bg + (size_t)(kb + 1) * 32);
      if (tid < 32) {
        tdm_issue(a0, g1);
        tdm_issue(b0, g1);
        __builtin_amdgcn_s_wait_tensorcnt(2);   // stage kb landed (in-order)
      }
    }
    __syncthreads();
    compute(kLdsA, kLdsB);
    __syncthreads();
    // ---- odd step: prefetch stage kb+2 -> buffer 0, consume buffer 1
    if (kb + 2 < nk) {
      const v4u a0 = tdm_g0(base + kLdsA, Ag + (size_t)(kb + 2) * 32);
      const v4u b0 = tdm_g0(base + kLdsB, Bg + (size_t)(kb + 2) * 32);
      if (tid < 32) {
        tdm_issue(a0, g1);
        tdm_issue(b0, g1);
        __builtin_amdgcn_s_wait_tensorcnt(2);   // stage kb+1 landed
      }
    } else if (tid < 32) {
      __builtin_amdgcn_s_wait_tensorcnt(0);     // final stage landed
    }
    __syncthreads();
    compute(kLdsA + kTileBytes, kLdsB + kTileBytes);
    __syncthreads();
  }
}

// ---------------------------------------------------------------- converts
__global__ __launch_bounds__(256)
void f32_to_f16_kernel(const float* __restrict__ in,
                       _Float16* __restrict__ out, int n) {
  int i = blockIdx.x * 256 + threadIdx.x;
  const int stride = gridDim.x * 256;
  for (; i < n; i += stride) out[i] = (_Float16)in[i];
}

// adjh = f16(adj + I): folds the "+outputs" term into the GEMM.
__global__ __launch_bounds__(256)
void conv_adj_kernel(const float* __restrict__ in,
                     _Float16* __restrict__ out, int n) {
  int i = blockIdx.x * 256 + threadIdx.x;
  const int stride = gridDim.x * 256;
  for (; i < n; i += stride) {
    float v = in[i];
    const int rc = i & (kS * kS - 1);
    if ((rc >> 9) == (rc & (kS - 1))) v += 1.0f;
    out[i] = (_Float16)v;
  }
}

// rden[row] = 1 / (sum_t adj[row][t] + 1), one wave32 per row.
__global__ __launch_bounds__(256)
void denom_kernel(const float* __restrict__ adj, float* __restrict__ rden) {
  const int row  = blockIdx.x * 8 + (threadIdx.x >> 5);
  const int lane = threadIdx.x & 31;
  const float* p = adj + (size_t)row * kS;
  float s = 0.f;
  for (int i = lane; i < kS; i += 32) s += p[i];
  for (int off = 16; off > 0; off >>= 1) s += __shfl_xor(s, off, 32);
  if (lane == 0) rden[row] = 1.0f / (s + 1.0f);
}

// [B,S,D] f16 -> [B,D,S] f16, 32x32 LDS tiles.
__global__ __launch_bounds__(256)
void transpose_kernel(const _Float16* __restrict__ X,
                      _Float16* __restrict__ Xt) {
  __shared__ _Float16 tile[32][33];
  const int b  = blockIdx.z;
  const int s0 = blockIdx.x * 32, d0 = blockIdx.y * 32;
  const _Float16* Xb = X + (size_t)b * kS * kD;
  _Float16* Xtb      = Xt + (size_t)b * kD * kS;
  const int tx = threadIdx.x & 31, ty = threadIdx.x >> 5;
  for (int r = ty; r < 32; r += 8) tile[r][tx] = Xb[(size_t)(s0 + r) * kD + d0 + tx];
  __syncthreads();
  for (int r = ty; r < 32; r += 8) Xtb[(size_t)(d0 + r) * kS + s0 + tx] = tile[tx][r];
}

// ------------------------------------- GEMM1: T = (adj + I) @ X  (per batch)
__global__ __launch_bounds__(256)
void gemm1_kernel(const _Float16* __restrict__ adjh,  // [B,S,S] (this head, +I)
                  const _Float16* __restrict__ Xt,    // [B,D,S]
                  _Float16* __restrict__ T) {         // [B,S,D]
  __shared__ __align__(128) char sm[kLdsTot];
  const int b  = blockIdx.z;
  const int mB = blockIdx.x * 128, nB = blockIdx.y * 128;
  const _Float16* Ag = adjh + (size_t)b * kS * kS + (size_t)mB * kS;
  const _Float16* Bg = Xt + (size_t)b * kD * kS + (size_t)nB * kS;
  _Float16* Tb = T + (size_t)b * kS * kD;

  v8f acc[2][4] = {};
  gemm_tdm_core(Ag, Bg, kS, kS / 32, sm, acc);

  const int wave = threadIdx.x >> 5;
  const int m0 = mB + ((wave >> 1) << 5);
  const int n0 = nB + ((wave & 1) << 6);
  const int lane = threadIdx.x & 31;
  const int col = lane & 15, rb = (lane >> 4) << 3;
#pragma unroll
  for (int mi = 0; mi < 2; ++mi)
#pragma unroll
    for (int ni = 0; ni < 4; ++ni)
#pragma unroll
      for (int v = 0; v < 8; ++v)
        Tb[(size_t)(m0 + mi * 16 + rb + v) * kD + n0 + ni * 16 + col] =
            (_Float16)acc[mi][ni][v];
}

// ------------- GEMM2: X' = relu((T @ W^T + 2b) * rden); also writes concat
__global__ __launch_bounds__(256)
void gemm2_kernel(const _Float16* __restrict__ T,    // [M, D]
                  const _Float16* __restrict__ W,    // [D, D] (row = out dim)
                  const float* __restrict__ bias,    // [D]
                  const float* __restrict__ rden,    // [M] (this head)
                  _Float16* __restrict__ Xn,         // [M, D]
                  _Float16* __restrict__ fin,        // [M, F]
                  int foff) {
  __shared__ __align__(128) char sm[kLdsTot];
  const int mB = blockIdx.x * 128, nB = blockIdx.y * 128;
  v8f acc[2][4] = {};
  gemm_tdm_core(T + (size_t)mB * kD, W + (size_t)nB * kD, kD, kD / 32, sm, acc);

  const int wave = threadIdx.x >> 5;
  const int m0 = mB + ((wave >> 1) << 5);
  const int n0 = nB + ((wave & 1) << 6);
  const int lane = threadIdx.x & 31;
  const int col = lane & 15, rb = (lane >> 4) << 3;
#pragma unroll
  for (int mi = 0; mi < 2; ++mi)
#pragma unroll
    for (int ni = 0; ni < 4; ++ni) {
      const int n = n0 + ni * 16 + col;
      const float bval = 2.0f * bias[n];
#pragma unroll
      for (int v = 0; v < 8; ++v) {
        const int m = m0 + mi * 16 + rb + v;
        float z = (acc[mi][ni][v] + bval) * rden[m];
        z = z > 0.0f ? z : 0.0f;
        const _Float16 h = (_Float16)z;
        Xn[(size_t)m * kD + n] = h;
        fin[(size_t)m * kF + foff + n] = h;
      }
    }
}

// ---------------- GEMM3: out = gcn + final @ Wout^T + b_out (f32 output)
__global__ __launch_bounds__(256)
void gemm3_kernel(const _Float16* __restrict__ fin,   // [M, F]
                  const _Float16* __restrict__ Wout,  // [D, F]
                  const float* __restrict__ bout,     // [D]
                  const float* __restrict__ gcn,      // [M, D]
                  float* __restrict__ out) {          // [M, D]
  __shared__ __align__(128) char sm[kLdsTot];
  const int mB = blockIdx.x * 128, nB = blockIdx.y * 128;
  v8f acc[2][4] = {};
  gemm_tdm_core(fin + (size_t)mB * kF, Wout + (size_t)nB * kF, kF,
                kF / 32, sm, acc);

  const int wave = threadIdx.x >> 5;
  const int m0 = mB + ((wave >> 1) << 5);
  const int n0 = nB + ((wave & 1) << 6);
  const int lane = threadIdx.x & 31;
  const int col = lane & 15, rb = (lane >> 4) << 3;
#pragma unroll
  for (int mi = 0; mi < 2; ++mi)
#pragma unroll
    for (int ni = 0; ni < 4; ++ni) {
      const int n = n0 + ni * 16 + col;
      const float bv = bout[n];
#pragma unroll
      for (int v = 0; v < 8; ++v) {
        const int m = m0 + mi * 16 + rb + v;
        out[(size_t)m * kD + n] = gcn[(size_t)m * kD + n] + acc[mi][ni][v] + bv;
      }
    }
}

} // namespace

extern "C" void kernel_launch(void* const* d_in, const int* in_sizes, int n_in,
                              void* d_out, int out_size, void* d_ws, size_t ws_size,
                              hipStream_t stream) {
  const float* adj  = (const float*)d_in[0];  // [H,B,S,S]
  const float* gcn  = (const float*)d_in[1];  // [B,S,D]
  // d_in[2] mask, d_in[3] domain_mask: unused by the forward pass
  const float* Wg   = (const float*)d_in[4];  // [H*L, D, D]
  const float* bg   = (const float*)d_in[5];  // [H*L, D]
  const float* Wout = (const float*)d_in[6];  // [D, F]
  const float* bout = (const float*)d_in[7];  // [D]
  float* out = (float*)d_out;

  char* ws = (char*)d_ws;
  size_t off = 0;
  auto carve = [&](size_t bytes) -> void* {
    void* p = ws + off;
    off += (bytes + 255) & ~(size_t)255;
    return p;
  };
  _Float16* adjh  = (_Float16*)carve((size_t)kH * kB * kS * kS * 2); // 25.2 MB
  _Float16* X0h   = (_Float16*)carve((size_t)kM * kD * 2);           //  8.4 MB
  _Float16* Xa    = (_Float16*)carve((size_t)kM * kD * 2);           //  8.4 MB
  _Float16* Xt    = (_Float16*)carve((size_t)kM * kD * 2);           //  8.4 MB
  _Float16* T     = (_Float16*)carve((size_t)kM * kD * 2);           //  8.4 MB
  _Float16* fin   = (_Float16*)carve((size_t)kM * kF * 2);           // 50.3 MB
  _Float16* Wgh   = (_Float16*)carve((size_t)kH * kL * kD * kD * 2); //  3.1 MB
  _Float16* Wouth = (_Float16*)carve((size_t)kD * kF * 2);           //  3.1 MB
  float*    rden  = (float*)carve((size_t)kH * kB * kS * 4);         //  0.1 MB
  (void)ws_size; (void)in_sizes; (void)n_in; (void)out_size;

  const dim3 blk(256);
  auto cgrid = [](int n) { int b = (n + 255) / 256; return dim3(b > 4096 ? 4096 : b); };

  conv_adj_kernel<<<cgrid(kH * kB * kS * kS), blk, 0, stream>>>(adj, adjh, kH * kB * kS * kS);
  f32_to_f16_kernel<<<cgrid(kM * kD), blk, 0, stream>>>(gcn, X0h, kM * kD);
  f32_to_f16_kernel<<<cgrid(kH * kL * kD * kD), blk, 0, stream>>>(Wg, Wgh, kH * kL * kD * kD);
  f32_to_f16_kernel<<<cgrid(kD * kF), blk, 0, stream>>>(Wout, Wouth, kD * kF);
  denom_kernel<<<dim3(kH * kB * kS / 8), blk, 0, stream>>>(adj, rden);

  const dim3 grid_t(kS / 32, kD / 32, kB);
  const dim3 grid_g1(kS / 128, kD / 128, kB);
  const dim3 grid_g2(kM / 128, kD / 128, 1);

  for (int h = 0; h < kH; ++h) {
    transpose_kernel<<<grid_t, blk, 0, stream>>>(X0h, Xt);
    for (int l = 0; l < kL; ++l) {
      const int idx = h * kL + l;
      gemm1_kernel<<<grid_g1, blk, 0, stream>>>(
          adjh + (size_t)h * kB * kS * kS, Xt, T);
      gemm2_kernel<<<grid_g2, blk, 0, stream>>>(
          T, Wgh + (size_t)idx * kD * kD, bg + (size_t)idx * kD,
          rden + (size_t)h * kB * kS, Xa, fin, idx * kD);
      if (l + 1 < kL)
        transpose_kernel<<<grid_t, blk, 0, stream>>>(Xa, Xt);
    }
  }

  gemm3_kernel<<<grid_g2, blk, 0, stream>>>(fin, Wouth, bout, gcn, out);
}
